// Box3dAttention_231928234562
// MI455X (gfx1250) — matrix operand order repaired and checked by reading, not compiled, not run on hardware
//
#include <hip/hip_runtime.h>
#include <hip/hip_bf16.h>

#define D_MODEL   256
#define NUM_HEAD  8
#define NUM_POINT 25
#define NUM_VAR   5
#define HEAD_DIM  32
#define NA        200   // NUM_HEAD * NUM_LEVEL * NUM_POINT
#define NB        40    // NUM_LEVEL * NUM_HEAD * NUM_VAR
#define LQ_CONST  4096
#define KCHUNKS   (D_MODEL / 32)          // 8
#define TILE_DW   (KCHUNKS * 32 * 8)      // dwords per packed B tile = 2048

typedef __attribute__((ext_vector_type(16))) __bf16 v16bf;
typedef __attribute__((ext_vector_type(8)))  float  v8f;

// One-instruction pack: truncate two fp32 to bf16 {hi[31:16], lo[31:16]} via v_perm_b32.
__device__ __forceinline__ unsigned pack2_bf16_trunc(float lo, float hi) {
  return __builtin_amdgcn_perm(__float_as_uint(hi), __float_as_uint(lo), 0x07060302u);
}

// RNE pack (used only in the one-time weight prep).
__device__ __forceinline__ unsigned pack2_bf16_rne(float lo, float hi) {
  unsigned a = __float_as_uint(lo);
  unsigned b = __float_as_uint(hi);
  a += 0x7FFFu + ((a >> 16) & 1u);
  b += 0x7FFFu + ((b >> 16) & 1u);
  return (a >> 16) | (b & 0xFFFF0000u);
}

// ---------------------------------------------------------------------------
// Pre-pack W (O x 256, fp32, row-major) into bf16 B-fragment order, zero-padded
// to colTiles4 = ceil(O/16) rounded up to multiple of 4 column tiles.
// dword address: tile*2048 + chunk*256 + lane*8 + v
//   lane: n = lane&15 (output col within tile), khalf = lane>>4
//   v:    k = chunk*32 + khalf*16 + 2v  (bf16 pair k, k+1)
// ---------------------------------------------------------------------------
__global__ void __launch_bounds__(256)
pack_weights_bf16(const float* __restrict__ W, unsigned* __restrict__ Wb,
                  int O, int totalDwords)
{
  const int t = blockIdx.x * 256 + threadIdx.x;
  if (t >= totalDwords) return;
  const int v     = t & 7;
  const int lane  = (t >> 3) & 31;
  const int chunk = (t >> 8) & (KCHUNKS - 1);
  const int tile  = t >> 11;
  const int n = lane & 15, khalf = lane >> 4;
  const int k = chunk * 32 + khalf * 16 + 2 * v;
  const int o = tile * 16 + n;
  unsigned r = 0u;
  if (o < O) {
    const float* p = W + (size_t)o * D_MODEL + k;
    r = pack2_bf16_rne(p[0], p[1]);
  }
  Wb[t] = r;
}

// ---------------------------------------------------------------------------
// Y[n,o] = sum_k X[n,k] * W[o,k] + bias[o], K = 256, N % 16 == 0.
// One wave computes a 16x64 strip: 4 column tiles, 4 f32 accumulators.
// Per K-chunk: build A fragment once (v_perm pack), load ALL 4 B fragments
// into distinct registers (one clause of 8 b128 loads) so the loadcnt waits
// stagger across the 4 WMMAs instead of serializing load->wait0->wmma.
// ---------------------------------------------------------------------------
__global__ void __launch_bounds__(256)
gemm_bias_bf16wmma(const float* __restrict__ X, const unsigned* __restrict__ Wb,
                   const float* __restrict__ bias, float* __restrict__ Y,
                   int N, int O, int colGroups)
{
  const int wave = blockIdx.x * (blockDim.x >> 5) + (threadIdx.x >> 5);
  const int rowTile = wave / colGroups;
  const int cg      = wave - rowTile * colGroups;
  if (rowTile * 16 >= N) return;                  // wave-uniform exit, EXEC stays full

  const int lane = threadIdx.x & 31;
  const int half = lane >> 4;
  const int l16  = lane & 15;

  const float*    Abase = X  + (size_t)(rowTile * 16 + l16) * D_MODEL + 16 * half;
  const unsigned* Bbase = Wb + (size_t)(cg * 4) * TILE_DW + lane * 8;

  v8f acc[4] = {};
  #pragma unroll 2
  for (int kc = 0; kc < KCHUNKS; ++kc) {
    const float* Ak = Abase + kc * 32;
    if (kc + 1 < KCHUNKS) __builtin_prefetch(Ak + 32, 0, 3);   // WGP-scope prefetch of next A slab

    // Issue all B-fragment loads first (distinct registers -> overlapping loads).
    union { v16bf v; uint4 q[2]; } bf[4];
    #pragma unroll
    for (int t = 0; t < 4; ++t) {
      const unsigned* bp = Bbase + (size_t)t * TILE_DW + kc * 256;
      bf[t].q[0] = *(const uint4*)(bp);
      bf[t].q[1] = *(const uint4*)(bp + 4);
    }

    union { v16bf v; unsigned u[8]; } af;
    #pragma unroll
    for (int v = 0; v < 8; ++v) {
      const int kb = 2 * v;                                    // 0,2,4,...,14
      float2 p = *(const float2*)(Ak + kb);
      af.u[v] = pack2_bf16_trunc(p.x, p.y);                    // 1 v_perm_b32 each
    }

    #pragma unroll
    for (int t = 0; t < 4; ++t) {
      acc[t] = __builtin_amdgcn_wmma_f32_16x16x32_bf16(false, af.v, false, bf[t].v,
                                                       (short)0, acc[t], false, false);
    }
  }

  #pragma unroll
  for (int t = 0; t < 4; ++t) {
    const int ocol = (cg * 4 + t) * 16 + l16;
    if (ocol < O) {
      const float bs = bias[ocol];
      #pragma unroll
      for (int r = 0; r < 8; ++r) {
        const int row = rowTile * 16 + r + 8 * half;
        Y[(size_t)row * O + ocol] = acc[t][r] + bs;
      }
    }
  }
}

// Per-(flat query, head) softmax over 25 logits -> straight into d_out attn region.
__global__ void __launch_bounds__(256)
softmax25_kernel(const float* __restrict__ logits, float* __restrict__ probs, int nRows)
{
  const int r = blockIdx.x * blockDim.x + threadIdx.x;   // r = flatq*8 + h
  if (r >= nRows) return;
  const int q = r >> 3, h = r & 7;
  const float* in  = logits + (size_t)q * NA + h * NUM_POINT;
  float*       out = probs  + (size_t)q * NA + h * NUM_POINT;
  float buf[NUM_POINT];
  float m = -3.4e38f;
  #pragma unroll
  for (int p = 0; p < NUM_POINT; ++p) { buf[p] = in[p]; m = fmaxf(m, buf[p]); }
  float s = 0.0f;
  #pragma unroll
  for (int p = 0; p < NUM_POINT; ++p) { buf[p] = __expf(buf[p] - m); s += buf[p]; }
  const float inv = 1.0f / s;
  #pragma unroll
  for (int p = 0; p < NUM_POINT; ++p) out[p] = buf[p] * inv;
}

// One wave per (flat query, head); lane = head-dim channel (HEAD_DIM == 32 == wave32).
__global__ void __launch_bounds__(256)
box3d_sample_kernel(const float* __restrict__ vproj,
                    const float* __restrict__ ob,
                    const float* __restrict__ refw,
                    const float* __restrict__ attn,
                    float* __restrict__ out_heads,
                    const int* __restrict__ Hfp, const int* __restrict__ Wfp,
                    int NQ, int LQ)
{
  const int wave = blockIdx.x * (blockDim.x >> 5) + (threadIdx.x >> 5);
  const int lane = threadIdx.x & 31;
  if (wave >= NQ * NUM_HEAD) return;
  const int q = wave >> 3;
  const int h = wave & 7;
  const int b = q / LQ;
  const int Hf = *Hfp, Wf = *Wfp;
  const int LV = Hf * Wf;

  const float* rw = refw + (size_t)q * 7;
  const float* o  = ob   + (size_t)q * NB + h * NUM_VAR;
  const float cxr = rw[0], cyr = rw[1], wr = rw[3], hr = rw[4], ar = rw[6];
  const float cx = cxr + o[0] * 0.125f * wr;
  const float cy = cyr + o[1] * 0.125f * hr;
  const float bw = wr + o[2] * 0.125f * wr;
  const float bh = hr + o[3] * 0.125f * hr;
  const float ang = (ar + o[4] * 0.0625f) * 6.283185307179586f;
  const float ca = __cosf(ang), sa = __sinf(ang);
  const float sw = fmaxf(bw, 0.0f), sh = fmaxf(bh, 0.0f);

  const float* vb = vproj + (size_t)b * LV * D_MODEL + h * HEAD_DIM + lane;
  const float* ap = attn + (size_t)q * NA + h * NUM_POINT;

  auto corner = [&](int yi, int xi) -> float {
    const bool valid = (xi >= 0) & (xi < Wf) & (yi >= 0) & (yi < Hf);
    const int xc = min(max(xi, 0), Wf - 1);
    const int yc = min(max(yi, 0), Hf - 1);
    const float val = vb[(size_t)(yc * Wf + xc) * D_MODEL];
    return valid ? val : 0.0f;
  };

  float acc = 0.0f;
  for (int p = 0; p < NUM_POINT; ++p) {
    const int pi = p / 5, pj = p - pi * 5;
    const float gx0 = (float)(pj - 2) * 0.2f * sw;
    const float gy0 = (float)(pi - 2) * 0.2f * sh;
    const float gx = cx + gx0 * ca - gy0 * sa;
    const float gy = cy + gx0 * sa + gy0 * ca;
    const float px = gx * (float)Wf - 0.5f;
    const float py = gy * (float)Hf - 0.5f;
    const float x0f = floorf(px), y0f = floorf(py);
    const int x0 = (int)x0f, y0 = (int)y0f;
    const float wx1 = px - x0f, wy1 = py - y0f;
    const float wx0 = 1.0f - wx1, wy0 = 1.0f - wy1;
    const float s = corner(y0,     x0    ) * (wy0 * wx0)
                  + corner(y0,     x0 + 1) * (wy0 * wx1)
                  + corner(y0 + 1, x0    ) * (wy1 * wx0)
                  + corner(y0 + 1, x0 + 1) * (wy1 * wx1);
    acc += ap[p] * s;
  }
  out_heads[(size_t)q * D_MODEL + h * HEAD_DIM + lane] = acc;
}

// ---------------------------------------------------------------------------
// Host-side helpers
// ---------------------------------------------------------------------------
static inline int col_tiles4(int O) { return (((O + 15) / 16) + 3) & ~3; }

static inline void launch_pack(const float* W, unsigned* Wb, int O, hipStream_t stream) {
  const int total = col_tiles4(O) * TILE_DW;
  hipLaunchKernelGGL(pack_weights_bf16, dim3((total + 255) / 256), dim3(256), 0, stream,
                     W, Wb, O, total);
}

static inline void launch_gemm(const float* X, const unsigned* Wb, const float* bias,
                               float* Y, int N, int O, hipStream_t stream)
{
  const int rowTiles  = N / 16;
  const int colGroups = col_tiles4(O) / 4;
  const int waves  = rowTiles * colGroups;
  const int blocks = (waves + 7) / 8;   // 8 waves / 256-thread block
  hipLaunchKernelGGL(gemm_bias_bf16wmma, dim3(blocks), dim3(256), 0, stream,
                     X, Wb, bias, Y, N, O, colGroups);
}

extern "C" void kernel_launch(void* const* d_in, const int* in_sizes, int n_in,
                              void* d_out, int out_size, void* d_ws, size_t ws_size,
                              hipStream_t stream)
{
  const float* query   = (const float*)d_in[0];
  const float* value   = (const float*)d_in[1];
  const float* refw    = (const float*)d_in[2];
  const float* W_box   = (const float*)d_in[3];
  const float* b_box   = (const float*)d_in[4];
  const float* W_attn  = (const float*)d_in[5];
  const float* b_attn  = (const float*)d_in[6];
  const float* W_value = (const float*)d_in[7];
  const float* b_value = (const float*)d_in[8];
  const float* W_out   = (const float*)d_in[9];
  const float* b_out   = (const float*)d_in[10];
  const int*   H_feat  = (const int*)d_in[11];
  const int*   W_feat  = (const int*)d_in[12];

  const int NQ  = in_sizes[0] / D_MODEL;   // B * LQ  (8192)
  const int NVr = in_sizes[1] / D_MODEL;   // B * LV  (70688)
  const int LQ  = LQ_CONST;

  // Workspace layout
  float* ws        = (float*)d_ws;
  float* v_proj    = ws;                                    // NVr * 256  (~72 MB, L2-resident)
  float* logits    = v_proj    + (size_t)NVr * D_MODEL;     // NQ * 200
  float* ob        = logits    + (size_t)NQ * NA;           // NQ * 40
  float* out_heads = ob        + (size_t)NQ * NB;           // NQ * 256
  unsigned* wv_b   = (unsigned*)(out_heads + (size_t)NQ * D_MODEL);
  unsigned* wa_b   = wv_b   + (size_t)col_tiles4(D_MODEL) * TILE_DW;
  unsigned* wbx_b  = wa_b   + (size_t)col_tiles4(NA)      * TILE_DW;
  unsigned* wo_b   = wbx_b  + (size_t)col_tiles4(NB)      * TILE_DW;

  float* out_ptr  = (float*)d_out;                          // NQ * 256
  float* attn_ptr = out_ptr + (size_t)NQ * D_MODEL;         // NQ * 200 (output #2)

  // 0) one-time bf16 weight packing (fragment-ordered, zero-padded)
  launch_pack(W_value, wv_b,  D_MODEL, stream);
  launch_pack(W_attn,  wa_b,  NA,      stream);
  launch_pack(W_box,   wbx_b, NB,      stream);
  launch_pack(W_out,   wo_b,  D_MODEL, stream);

  // 1) value projection (dominant, memory-bound GEMM)
  launch_gemm(value, wv_b, b_value, v_proj, NVr, D_MODEL, stream);

  // 2) attention logits
  launch_gemm(query, wa_b, b_attn, logits, NQ, NA, stream);

  // 3) box offsets
  launch_gemm(query, wbx_b, b_box, ob, NQ, NB, stream);

  // 4) softmax -> d_out attn region
  {
    const int nRows = NQ * NUM_HEAD;
    hipLaunchKernelGGL(softmax25_kernel, dim3((nRows + 255) / 256), dim3(256), 0, stream,
                       logits, attn_ptr, nRows);
  }

  // 5) box decode + bilinear sampling + attn-weighted head accumulation
  {
    const int waves = NQ * NUM_HEAD;
    hipLaunchKernelGGL(box3d_sample_kernel, dim3((waves + 7) / 8), dim3(256), 0, stream,
                       v_proj, ob, refw, attn_ptr, out_heads, H_feat, W_feat, NQ, LQ);
  }

  // 6) output projection -> d_out
  launch_gemm(out_heads, wo_b, b_out, out_ptr, NQ, D_MODEL, stream);
}
